// Net_31722628448714
// MI455X (gfx1250) — matrix operand order repaired and checked by reading, not compiled, not run on hardware
//
#include <hip/hip_runtime.h>
#include <math.h>

// ---------------------------------------------------------------------------
// Types for WMMA bf16 fragments (probe-confirmed builtin signature:
//   __builtin_amdgcn_wmma_f32_16x16x32_bf16(bool, v16bf, bool, v16bf, short, v8f, bool, bool)
// ---------------------------------------------------------------------------
typedef __attribute__((ext_vector_type(16))) __bf16 v16bf;
typedef __attribute__((ext_vector_type(8)))  float  v8f;

union Frag {
    v16bf f;
    uint4 q[2];
};

__device__ __forceinline__ unsigned short f32_to_bf16(float x) {
    unsigned u = __float_as_uint(x);
    unsigned r = u + 0x7FFFu + ((u >> 16) & 1u);   // round-to-nearest-even
    return (unsigned short)(r >> 16);
}

// Order-preserving float -> uint mapping for atomicMax-based segment max.
__device__ __forceinline__ unsigned f32_ord(float f) {
    unsigned u = __float_as_uint(f);
    return (u & 0x80000000u) ? ~u : (u | 0x80000000u);
}
__device__ __forceinline__ float ord_f32(unsigned u) {
    return (u & 0x80000000u) ? __uint_as_float(u & 0x7FFFFFFFu)
                             : __uint_as_float(~u);
}

// ---------------------------------------------------------------------------
// Elementwise converts
// ---------------------------------------------------------------------------
__global__ void k_f32_to_bf16(const float* __restrict__ in,
                              unsigned short* __restrict__ out, int n) {
    int i = blockIdx.x * blockDim.x + threadIdx.x;
    if (i < n) out[i] = f32_to_bf16(in[i]);
}

// W: [K, Ncol] row-major f32 -> Wt: [Ncol, K] bf16 (column-major B for GEMM)
__global__ void k_transpose_bf16(const float* __restrict__ W,
                                 unsigned short* __restrict__ Wt,
                                 int K, int Ncol) {
    int i = blockIdx.x * blockDim.x + threadIdx.x;
    if (i >= K * Ncol) return;
    int k = i / Ncol, n = i - k * Ncol;
    Wt[(size_t)n * K + k] = f32_to_bf16(W[i]);
}

// ---------------------------------------------------------------------------
// WMMA GEMM: C[M,Ncol] = A[M,256](bf16) * B[256,Ncol], with Bt = B^T [Ncol,256].
// Each wave computes one 16-row strip x NT consecutive 16-col tiles, reusing the
// A fragment across NT WMMAs per k-step. K is compile-time (256) so the inner
// loop is a branch-free stream of global_load_b128 clauses + v_wmma.
//
// Fragment layouts follow CDNA5 ISA 7.12.2:
//   A (16-bit 16x32): lane L -> row M=L%16; halves j: K = (j>>3)*16 + (L>=16?8:0) + (j&7)
//     => two contiguous 8-half (16B) runs per lane per k-step.
//   B (16-bit 32x16): lane L -> col N=L%16; halves j: K = (j>>3)*16 + (L>=16?8:0) + (j&7)
//     => two contiguous 16B runs per lane per k-step (from Bt).
//   C/D f32 16x16: lane L -> col N=L%16; VGPR r -> row M = (L>=16?8:0)+r.
// ---------------------------------------------------------------------------
template <int NT>
__global__ void k_wmma_gemm(const unsigned short* __restrict__ A,
                            const unsigned short* __restrict__ Bt,
                            float* __restrict__ C,
                            int M, int Ncol) {
    constexpr int K = 256;
    int wave = threadIdx.x >> 5;
    int lane = threadIdx.x & 31;
    int ngroups = Ncol / (16 * NT);
    int tile = blockIdx.x * (blockDim.x >> 5) + wave;
    int total = (M >> 4) * ngroups;
    if (tile >= total) return;            // uniform per wave: EXEC stays all-1s
    int mtile = tile / ngroups;
    int ngrp  = tile - mtile * ngroups;

    int mn = lane & 15;        // row within A tile / col within B tile
    int hi = lane >> 4;        // 0 or 1

    const unsigned short* arow  = A  + (size_t)(mtile * 16 + mn) * K + hi * 8;
    const unsigned short* bbase = Bt + (size_t)(ngrp * NT * 16 + mn) * K + hi * 8;

    v8f acc[NT];
#pragma unroll
    for (int t = 0; t < NT; ++t) acc[t] = (v8f){};

#pragma unroll
    for (int k0 = 0; k0 < K; k0 += 32) {
        Frag a;
        a.q[0] = *(const uint4*)(arow + k0);          // K = k0 + hi*8  + 0..7
        a.q[1] = *(const uint4*)(arow + k0 + 16);     // K = k0+16+hi*8 + 0..7
#pragma unroll
        for (int t = 0; t < NT; ++t) {
            const unsigned short* bcol = bbase + (size_t)t * 16 * K;
            Frag b;
            b.q[0] = *(const uint4*)(bcol + k0);      // K = k0 + hi*8  + 0..7
            b.q[1] = *(const uint4*)(bcol + k0 + 16); // K = k0+16+hi*8 + 0..7
            acc[t] = __builtin_amdgcn_wmma_f32_16x16x32_bf16(
                       false, a.f, false, b.f, (short)0, acc[t], false, false);
        }
    }

    int rowbase = mtile * 16 + hi * 8;
#pragma unroll
    for (int t = 0; t < NT; ++t) {
        int col = (ngrp * NT + t) * 16 + mn;
#pragma unroll
        for (int r = 0; r < 8; ++r)
            C[(size_t)(rowbase + r) * Ncol + col] = acc[t][r];
    }
}

// ---------------------------------------------------------------------------
// Attention coefficients per node+head, and per-launch init of emax/esum.
// xl: [N, H*C]; a_src/a_dst: [H, C]
// ---------------------------------------------------------------------------
__global__ void k_alpha(const float* __restrict__ xl,
                        const float* __restrict__ a_src,
                        const float* __restrict__ a_dst,
                        float* __restrict__ asrc, float* __restrict__ adst,
                        unsigned* __restrict__ emax, float* __restrict__ esum,
                        int N, int H, int C) {
    int i = blockIdx.x * blockDim.x + threadIdx.x;   // i = n*H + h
    if (i >= N * H) return;
    int h = i % H;
    const float* row = xl + (size_t)(i / H) * (H * C) + h * C;
    const float* as = a_src + h * C;
    const float* ad = a_dst + h * C;
    float s = 0.f, d = 0.f;
    for (int c = 0; c < C; ++c) { float v = row[c]; s += v * as[c]; d += v * ad[c]; }
    asrc[i] = s;
    adst[i] = d;
    emax[i] = 0u;       // encoded "-inf" floor (every node has a self-loop edge)
    esum[i] = 0.f;
}

// per-edge segment max (encoded uint atomicMax)
__global__ void k_edge_max(const int* __restrict__ ei,
                           const float* __restrict__ asrc,
                           const float* __restrict__ adst,
                           unsigned* __restrict__ emax, int E, int H) {
    int i = blockIdx.x * blockDim.x + threadIdx.x;   // i = e*H + h
    if (i >= E * H) return;
    int e = i / H, h = i - e * H;
    int s = ei[e], d = ei[E + e];
    float v = asrc[s * H + h] + adst[d * H + h];
    v = (v > 0.f) ? v : 0.2f * v;                    // leaky_relu(0.2)
    atomicMax(&emax[d * H + h], f32_ord(v));
}

// per-edge exp + segment sum; stores exp(e - max) into ee[e*H+h]
__global__ void k_edge_sum(const int* __restrict__ ei,
                           const float* __restrict__ asrc,
                           const float* __restrict__ adst,
                           const unsigned* __restrict__ emax,
                           float* __restrict__ esum,
                           float* __restrict__ ee, int E, int H) {
    int i = blockIdx.x * blockDim.x + threadIdx.x;
    if (i >= E * H) return;
    int e = i / H, h = i - e * H;
    int s = ei[e], d = ei[E + e];
    float v = asrc[s * H + h] + adst[d * H + h];
    v = (v > 0.f) ? v : 0.2f * v;
    float x = __expf(v - ord_f32(emax[d * H + h]));
    ee[i] = x;
    atomicAdd(&esum[d * H + h], x);
}

// message scatter: one thread per (edge, head, channel) -> coalesced atomics
__global__ void k_scatter(const int* __restrict__ ei,
                          const float* __restrict__ xl,
                          const float* __restrict__ ee,
                          const float* __restrict__ esum,
                          float* __restrict__ out, int E, int H, int C) {
    long long i = (long long)blockIdx.x * blockDim.x + threadIdx.x;
    long long total = (long long)E * H * C;
    if (i >= total) return;
    int c = (int)(i % C);
    long long t = i / C;
    int h = (int)(t % H);
    int e = (int)(t / H);
    int s = ei[e], d = ei[E + e];
    float alpha = ee[(size_t)e * H + h] / esum[d * H + h];
    atomicAdd(&out[(size_t)d * (H * C) + h * C + c],
              xl[(size_t)s * (H * C) + h * C + c] * alpha);
}

// layer-1 epilogue: +bias, ELU, convert to bf16 for the layer-2 GEMM A operand
__global__ void k_bias_elu_bf16(const float* __restrict__ agg,
                                const float* __restrict__ bias,
                                unsigned short* __restrict__ hb, int N) {
    int i = blockIdx.x * blockDim.x + threadIdx.x;
    if (i >= N * 256) return;
    float v = agg[i] + bias[i & 255];
    v = (v > 0.f) ? v : (__expf(v) - 1.f);
    hb[i] = f32_to_bf16(v);
}

// final: +b2, log_softmax over 32 channels
__global__ void k_logsoftmax(const float* __restrict__ agg,
                             const float* __restrict__ b2,
                             float* __restrict__ y, int N) {
    int n = blockIdx.x * blockDim.x + threadIdx.x;
    if (n >= N) return;
    float v[32];
    float mx = -1e30f;
#pragma unroll
    for (int c = 0; c < 32; ++c) {
        v[c] = agg[(size_t)n * 32 + c] + b2[c];
        mx = fmaxf(mx, v[c]);
    }
    float s = 0.f;
#pragma unroll
    for (int c = 0; c < 32; ++c) s += __expf(v[c] - mx);
    float ls = mx + __logf(s);
#pragma unroll
    for (int c = 0; c < 32; ++c) y[(size_t)n * 32 + c] = v[c] - ls;
}

// ---------------------------------------------------------------------------
// Launch
// ---------------------------------------------------------------------------
extern "C" void kernel_launch(void* const* d_in, const int* in_sizes, int n_in,
                              void* d_out, int out_size, void* d_ws, size_t ws_size,
                              hipStream_t stream) {
    (void)n_in; (void)out_size; (void)ws_size;
    const float* x   = (const float*)d_in[0];
    const int*   ei  = (const int*)  d_in[1];
    const float* W1  = (const float*)d_in[2];
    const float* b1  = (const float*)d_in[3];
    const float* as1 = (const float*)d_in[4];
    const float* ad1 = (const float*)d_in[5];
    const float* W2  = (const float*)d_in[6];
    const float* b2  = (const float*)d_in[7];
    const float* as2 = (const float*)d_in[8];
    const float* ad2 = (const float*)d_in[9];
    float* y = (float*)d_out;

    const int N = in_sizes[0] / 256;   // 50000
    const int E = in_sizes[1] / 2;     // 850000 (incl. self-loops)

    // workspace layout (256B-aligned regions)
    char* ws = (char*)d_ws;
    size_t off = 0;
    auto take = [&](size_t bytes) -> char* {
        char* p = ws + off;
        off += (bytes + 255) & ~(size_t)255;
        return p;
    };
    unsigned short* xb   = (unsigned short*)take((size_t)N * 256 * 2); // x bf16, reused as h bf16
    unsigned short* w1t  = (unsigned short*)take((size_t)256 * 256 * 2);
    unsigned short* w2t  = (unsigned short*)take((size_t)32 * 256 * 2);
    float*    asrc1 = (float*)   take((size_t)N * 8 * 4);
    float*    adst1 = (float*)   take((size_t)N * 8 * 4);
    unsigned* emax1 = (unsigned*)take((size_t)N * 8 * 4);
    float*    esum1 = (float*)   take((size_t)N * 8 * 4);
    float*    e1    = (float*)   take((size_t)E * 8 * 4);
    float*    out1  = (float*)   take((size_t)N * 256 * 4);
    float*    h1    = (float*)   take((size_t)N * 256 * 4);
    // layer-2 intermediates alias the h1 region (dead after layer-1 scatter)
    float*    h2     = h1;                      // N*32
    float*    asrc2  = h2 + (size_t)N * 32;     // N
    float*    adst2  = asrc2 + N;               // N
    unsigned* emax2  = (unsigned*)(adst2 + N);  // N
    float*    esum2  = (float*)(emax2 + N);     // N
    float*    e2     = esum2 + N;               // E
    float*    out2   = e2 + E;                  // N*32

    const int T = 256;

    // ---- precision conversion + weight transposes ----
    k_f32_to_bf16 <<<(N * 256 + T - 1) / T, T, 0, stream>>>(x, xb, N * 256);
    k_transpose_bf16<<<(256 * 256 + T - 1) / T, T, 0, stream>>>(W1, w1t, 256, 256);
    k_transpose_bf16<<<(256 * 32  + T - 1) / T, T, 0, stream>>>(W2, w2t, 256, 32);

    // ---- layer 1: xl = x @ W1  (WMMA bf16, NT=4 col-tiles per wave) ----
    {
        int total = (N / 16) * (256 / (16 * 4));          // 3125 * 4
        k_wmma_gemm<4><<<(total + 3) / 4, 128, 0, stream>>>(xb, w1t, h1, N, 256);
    }
    k_alpha<<<(N * 8 + T - 1) / T, T, 0, stream>>>(h1, as1, ad1,
                                                   asrc1, adst1, emax1, esum1, N, 8, 32);
    hipMemsetAsync(out1, 0, (size_t)N * 256 * 4, stream);
    k_edge_max<<<(E * 8 + T - 1) / T, T, 0, stream>>>(ei, asrc1, adst1, emax1, E, 8);
    k_edge_sum<<<(E * 8 + T - 1) / T, T, 0, stream>>>(ei, asrc1, adst1, emax1, esum1, e1, E, 8);
    {
        long long tot = (long long)E * 8 * 32;
        k_scatter<<<(int)((tot + T - 1) / T), T, 0, stream>>>(ei, h1, e1, esum1, out1, E, 8, 32);
    }
    // +b1, ELU, -> bf16 A operand for layer 2
    k_bias_elu_bf16<<<(N * 256 + T - 1) / T, T, 0, stream>>>(out1, b1, xb, N);

    // ---- layer 2: xl2 = h @ W2  (WMMA bf16, NT=2 col-tiles per wave) ----
    {
        int total = (N / 16) * (32 / (16 * 2));           // 3125
        k_wmma_gemm<2><<<(total + 3) / 4, 128, 0, stream>>>(xb, w2t, h2, N, 32);
    }
    k_alpha<<<(N + T - 1) / T, T, 0, stream>>>(h2, as2, ad2,
                                               asrc2, adst2, emax2, esum2, N, 1, 32);
    hipMemsetAsync(out2, 0, (size_t)N * 32 * 4, stream);
    k_edge_max<<<(E + T - 1) / T, T, 0, stream>>>(ei, asrc2, adst2, emax2, E, 1);
    k_edge_sum<<<(E + T - 1) / T, T, 0, stream>>>(ei, asrc2, adst2, emax2, esum2, e2, E, 1);
    {
        long long tot = (long long)E * 32;
        k_scatter<<<(int)((tot + T - 1) / T), T, 0, stream>>>(ei, h2, e2, esum2, out2, E, 1, 32);
    }
    k_logsoftmax<<<(N + T - 1) / T, T, 0, stream>>>(out2, b2, y, N);
}
